// TripletLoss_51110110822513
// MI455X (gfx1250) — compile-verified, matrix operand order checked by reading, not run on hardware
//
#include <hip/hip_runtime.h>
#include <hip/hip_bf16.h>
#include <math.h>

typedef __attribute__((ext_vector_type(16))) _Float16 v16h;
typedef __attribute__((ext_vector_type(8)))  _Float16 v8h;
typedef __attribute__((ext_vector_type(4)))  _Float16 v4h;
typedef __attribute__((ext_vector_type(8)))  float    v8f;

#define TL_MARGIN 0.3f

// ---------------------------------------------------------------------------
// init: zero pos_sum, +inf min_sh (as ordered uint bits), zero histogram
// ---------------------------------------------------------------------------
__global__ void k_init(float* __restrict__ pos_sum, unsigned* __restrict__ min_sh,
                       int* __restrict__ hist, int B) {
  int i = blockIdx.x * blockDim.x + threadIdx.x;
  if (i < B) { pos_sum[i] = 0.0f; min_sh[i] = 0x7f800000u; }
  if (i < 128) hist[i] = 0;
}

// ---------------------------------------------------------------------------
// normalize rows (f32 -> unit-norm f16 copy) + label histogram
// one block (D/4 threads) per row; D == 512 -> 128 threads = 4 waves
// ---------------------------------------------------------------------------
__global__ void k_normalize(const float* __restrict__ feat,
                            const int* __restrict__ labels,
                            _Float16* __restrict__ fh,
                            int* __restrict__ hist, int D) {
  int row = blockIdx.x;
  int t = threadIdx.x;
  const float4* src = (const float4*)(feat + (size_t)row * D);
  float4 v = src[t];
  float ss = v.x * v.x + v.y * v.y + v.z * v.z + v.w * v.w;
#pragma unroll
  for (int m = 1; m < 32; m <<= 1) ss += __shfl_xor(ss, m, 32);
  __shared__ float warp_ss[4];
  if ((t & 31) == 0) warp_ss[t >> 5] = ss;
  __syncthreads();
  float tot = warp_ss[0] + warp_ss[1] + warp_ss[2] + warp_ss[3];
  float scale = 1.0f / fmaxf(sqrtf(tot), 1e-12f);
  v4h o;
  o[0] = (_Float16)(v.x * scale);
  o[1] = (_Float16)(v.y * scale);
  o[2] = (_Float16)(v.z * scale);
  o[3] = (_Float16)(v.w * scale);
  *(v4h*)(fh + (size_t)row * D + (size_t)t * 4) = o;
  if (t == 0) atomicAdd(&hist[labels[row]], 1);
}

// ---------------------------------------------------------------------------
// WMMA fragment loader. Because B = f^T, A(16x32) and B(32x16) fragments are
// loaded identically from rows of the row-major f16 matrix:
//   lane & 15  -> M (for A) / N (for B)  == source row
//   lanes 0-15 hold K {0..7, 16..23}, lanes 16-31 hold K {8..15, 24..31}
// Each fragment = two contiguous 16-byte chunks -> two global_load_b128.
// ---------------------------------------------------------------------------
__device__ __forceinline__ v16h load_frag(const _Float16* __restrict__ fh,
                                          int row0, int kbase, int lane, int D) {
  int r  = row0 + (lane & 15);
  int kh = kbase + ((lane & 16) >> 1);     // hi half-wave: +8 halves
  const _Float16* p = fh + (size_t)r * D + kh;
  union { v16h v; v8h h[2]; } u;
  u.h[0] = *(const v8h*)(p);               // K kh..kh+7
  u.h[1] = *(const v8h*)(p + 16);          // K kh+16..kh+23
  return u.v;
}

#define WMMA_F16(a, b, c)                                                     \
  __builtin_amdgcn_wmma_f32_16x16x32_f16(false, (a), false, (b), (short)0,    \
                                         (c), false, false)

// Per-wave 2x(16x128) strip: 16 accumulator tiles; each B fragment feeds two
// WMMAs (acc0/acc1), halving B traffic per FLOP vs a 1x8 tile.
__device__ __forceinline__ void gemm_tile2(const _Float16* __restrict__ fh,
                                           int row0, int row1, int colbase,
                                           int D, int lane,
                                           v8f acc0[8], v8f acc1[8]) {
  for (int kb = 0; kb < D; kb += 32) {
    v16h a0 = load_frag(fh, row0, kb, lane, D);
    v16h a1 = load_frag(fh, row1, kb, lane, D);
#pragma unroll
    for (int ni = 0; ni < 8; ni++) {
      v16h b = load_frag(fh, colbase + ni * 16, kb, lane, D);
      acc0[ni] = WMMA_F16(a0, b, acc0[ni]);
      acc1[ni] = WMMA_F16(a1, b, acc1[ni]);
    }
  }
}

// ---------------------------------------------------------------------------
// pass-1 epilogue: masked sums of dist over positives for one 16x128 strip.
// Row anchors always; mirrored column anchors unless this is a diagonal tile.
// ---------------------------------------------------------------------------
__device__ __forceinline__ void epilogue_pos(const v8f acc[8], int rowbase,
                                             int colbase, bool diag,
                                             const int* __restrict__ labels,
                                             float* __restrict__ pos_sum,
                                             int lane) {
  int nlo = lane & 15;
  int hi  = (lane >> 4) & 1;
  int rlab[8];
#pragma unroll
  for (int r = 0; r < 8; r++) rlab[r] = labels[rowbase + 8 * hi + r];

  float rowsum[8] = {0, 0, 0, 0, 0, 0, 0, 0};
#pragma unroll
  for (int ni = 0; ni < 8; ni++) {
    int cg = colbase + ni * 16 + nlo;
    int clab = labels[cg];
    float colsum = 0.0f;
#pragma unroll
    for (int r = 0; r < 8; r++) {
      float g = acc[ni][r];
      float d = sqrtf(fmaxf(2.0f - 2.0f * g, 0.0f));
      int rg = rowbase + 8 * hi + r;
      if (rg == cg) d = 0.0f;                // exact self-distance
      if (rlab[r] == clab) { rowsum[r] += d; colsum += d; }
    }
    if (!diag) {                             // mirrored (col-anchor) contribution
      colsum += __shfl_xor(colsum, 16, 32);
      if (hi == 0) atomicAdd(&pos_sum[cg], colsum);
    }
  }
#pragma unroll
  for (int r = 0; r < 8; r++) {
    float v = rowsum[r];
#pragma unroll
    for (int m = 1; m < 16; m <<= 1) v += __shfl_xor(v, m, 32);
    if (nlo == 0) atomicAdd(&pos_sum[rowbase + 8 * hi + r], v);
  }
}

// ---------------------------------------------------------------------------
// pass-2 epilogue: min over semi-hard negatives for one 16x128 strip.
// ---------------------------------------------------------------------------
__device__ __forceinline__ void epilogue_min(const v8f acc[8], int rowbase,
                                             int colbase, bool diag,
                                             const int* __restrict__ labels,
                                             const float* __restrict__ mean_pos,
                                             unsigned* __restrict__ min_sh,
                                             int lane) {
  int nlo = lane & 15;
  int hi  = (lane >> 4) & 1;
  const float INF = __uint_as_float(0x7f800000u);

  int rlab[8]; float rmean[8], rmin[8];
#pragma unroll
  for (int r = 0; r < 8; r++) {
    int rg = rowbase + 8 * hi + r;
    rlab[r] = labels[rg];
    rmean[r] = mean_pos[rg];
    rmin[r] = INF;
  }
#pragma unroll
  for (int ni = 0; ni < 8; ni++) {
    int cg = colbase + ni * 16 + nlo;
    int clab = labels[cg];
    float cmean = mean_pos[cg];
    float cmin = INF;
#pragma unroll
    for (int r = 0; r < 8; r++) {
      float g = acc[ni][r];
      float d = sqrtf(fmaxf(2.0f - 2.0f * g, 0.0f));
      if (rlab[r] != clab) {
        if (d > rmean[r]) rmin[r] = fminf(rmin[r], d);
        if (d > cmean)    cmin    = fminf(cmin, d);
      }
    }
    if (!diag) {
      cmin = fminf(cmin, __shfl_xor(cmin, 16, 32));
      if (hi == 0) atomicMin(&min_sh[cg], __float_as_uint(cmin));
    }
  }
#pragma unroll
  for (int r = 0; r < 8; r++) {
    float v = rmin[r];
#pragma unroll
    for (int m = 1; m < 16; m <<= 1) v = fminf(v, __shfl_xor(v, m, 32));
    if (nlo == 0) atomicMin(&min_sh[rowbase + 8 * hi + r], __float_as_uint(v));
  }
}

// ---------------------------------------------------------------------------
// pass 1: masked sum of distances over positives -> pos_sum.
// Macro-tile 256(M) x 128(N); in 128-row tile units block (by,bx) covers
// row-tiles {2by, 2by+1} x col-tile bx; launch only bx >= 2by (triangle).
// ---------------------------------------------------------------------------
__global__ __launch_bounds__(256)
void k_pass1(const _Float16* __restrict__ fh, const int* __restrict__ labels,
             float* __restrict__ pos_sum, int B, int D) {
  int bx = blockIdx.x, by = blockIdx.y;
  if (bx < 2 * by) return;
  int lane = threadIdx.x & 31;
  int wave = threadIdx.x >> 5;
  int row0 = by * 256 + wave * 16;          // row-tile 2by
  int row1 = row0 + 128;                    // row-tile 2by+1
  int colbase = bx * 128;

  v8f acc0[8] = {}, acc1[8] = {};
  gemm_tile2(fh, row0, row1, colbase, D, lane, acc0, acc1);

  epilogue_pos(acc0, row0, colbase, bx == 2 * by, labels, pos_sum, lane);
  if (bx >= 2 * by + 1)                     // half 1 mirror handled elsewhere if bx==2by
    epilogue_pos(acc1, row1, colbase, bx == 2 * by + 1, labels, pos_sum, lane);
}

// ---------------------------------------------------------------------------
// mean_pos[i] = pos_sum[i] / count(label[i])
// ---------------------------------------------------------------------------
__global__ void k_mean(const int* __restrict__ labels,
                       const float* __restrict__ pos_sum,
                       const int* __restrict__ hist,
                       float* __restrict__ mean_pos, int B) {
  int i = blockIdx.x * blockDim.x + threadIdx.x;
  if (i >= B) return;
  mean_pos[i] = pos_sum[i] / (float)hist[labels[i]];
}

// ---------------------------------------------------------------------------
// pass 2: min over semi-hard negatives (dist > mean_pos, different label).
// ---------------------------------------------------------------------------
__global__ __launch_bounds__(256)
void k_pass2(const _Float16* __restrict__ fh, const int* __restrict__ labels,
             const float* __restrict__ mean_pos, unsigned* __restrict__ min_sh,
             int B, int D) {
  int bx = blockIdx.x, by = blockIdx.y;
  if (bx < 2 * by) return;
  int lane = threadIdx.x & 31;
  int wave = threadIdx.x >> 5;
  int row0 = by * 256 + wave * 16;
  int row1 = row0 + 128;
  int colbase = bx * 128;

  v8f acc0[8] = {}, acc1[8] = {};
  gemm_tile2(fh, row0, row1, colbase, D, lane, acc0, acc1);

  epilogue_min(acc0, row0, colbase, bx == 2 * by, labels, mean_pos, min_sh, lane);
  if (bx >= 2 * by + 1)
    epilogue_min(acc1, row1, colbase, bx == 2 * by + 1, labels, mean_pos, min_sh, lane);
}

// ---------------------------------------------------------------------------
// finalize: mean over contributing anchors -> scalar loss
// ---------------------------------------------------------------------------
__global__ void k_final(const int* __restrict__ labels, const int* __restrict__ hist,
                        const float* __restrict__ mean_pos,
                        const unsigned* __restrict__ min_sh,
                        float* __restrict__ out, int B) {
  __shared__ float ssum[256];
  __shared__ int   scnt[256];
  int t = threadIdx.x;
  float s = 0.0f; int c = 0;
  for (int i = t; i < B; i += 256) {
    int cnt = hist[labels[i]];
    unsigned mb = min_sh[i];
    if (cnt > 1 && cnt < B && mb < 0x7f800000u) {
      float per = mean_pos[i] - __uint_as_float(mb) + TL_MARGIN;
      s += fmaxf(per, 0.0f);
      c += 1;
    }
  }
  ssum[t] = s; scnt[t] = c;
  __syncthreads();
  for (int off = 128; off > 0; off >>= 1) {
    if (t < off) { ssum[t] += ssum[t + off]; scnt[t] += scnt[t + off]; }
    __syncthreads();
  }
  if (t == 0) out[0] = (scnt[0] > 0) ? ssum[0] / (float)scnt[0] : 0.0f;
}

// ---------------------------------------------------------------------------
extern "C" void kernel_launch(void* const* d_in, const int* in_sizes, int n_in,
                              void* d_out, int out_size, void* d_ws, size_t ws_size,
                              hipStream_t stream) {
  const float* feat  = (const float*)d_in[0];
  const int*  labels = (const int*)d_in[1];
  int B = in_sizes[1];                     // 8192
  int D = in_sizes[0] / B;                 // 512

  // workspace layout (~8.1 MB total)
  char* ws = (char*)d_ws;
  _Float16* fh = (_Float16*)ws;
  size_t off = ((size_t)B * D * sizeof(_Float16) + 255) & ~(size_t)255;
  float*    pos_sum  = (float*)(ws + off);     off += (size_t)B * sizeof(float);
  float*    mean_pos = (float*)(ws + off);     off += (size_t)B * sizeof(float);
  unsigned* min_sh   = (unsigned*)(ws + off);  off += (size_t)B * sizeof(unsigned);
  int*      hist     = (int*)(ws + off);

  k_init<<<(B + 255) / 256, 256, 0, stream>>>(pos_sum, min_sh, hist, B);
  k_normalize<<<B, D / 4, 0, stream>>>(feat, labels, fh, hist, D);

  dim3 grid(B / 128, B / 256);
  k_pass1<<<grid, 256, 0, stream>>>(fh, labels, pos_sum, B, D);
  k_mean<<<(B + 255) / 256, 256, 0, stream>>>(labels, pos_sum, hist, mean_pos, B);
  k_pass2<<<grid, 256, 0, stream>>>(fh, labels, mean_pos, min_sh, B, D);
  k_final<<<1, 256, 0, stream>>>(labels, hist, mean_pos, min_sh, (float*)d_out, B);
}
